// Model_77446850282174
// MI455X (gfx1250) — compile-verified, matrix operand order checked by reading, not run on hardware
//
#include <hip/hip_runtime.h>
#include <math.h>

// ---------------- problem constants ----------------
#define FEAT 161
#define HID  176
#define SEQ  456
#define BATCH 512

#define FP   176   // FEAT padded to N multiple of 16
#define KF   192   // 161 padded to K multiple of 32
#define KH   192   // 176 padded to K multiple of 32
#define K2   352   // 322 padded to K multiple of 32
#define G4   704   // 4*HID (exact multiple of 32 and 16)

#define MT   16            // batch rows per workgroup (one WMMA M tile)
#define NWG  (BATCH / MT)  // 32 workgroups
#define NTHREADS 256       // 8 waves of 32 (2 per SIMD32 -> latency hiding)
#define NWAVES 8

// ---------------- ws layout (bytes, all multiples of 256) ----------------
#define OFF_WDH   0u            // [176][192] f16
#define OFF_WHR   67584u        // [176][192] f16 (161 rows real)
#define OFF_WFR   135168u       // [176][192] f16 (off-diag masked)
#define OFF_WWC   202752u       // [176][352] f16
#define OFF_WIH   326656u       // [704][352] f16
#define OFF_WHH   822272u       // [704][192] f16
#define OFF_WIH2  1092608u      // [704][192] f16
#define OFF_WHH2  1362944u      // [704][192] f16
#define OFF_PART  1633280u      // float part[4][SEQ][NWG]

typedef __attribute__((ext_vector_type(16))) _Float16 v16h;
typedef __attribute__((ext_vector_type(8)))  _Float16 v8h;
typedef __attribute__((ext_vector_type(8)))  float    v8f;

// ---------------- weight convert + pad to f16 ----------------
__global__ void cvt_pad_f16(const float* __restrict__ src, _Float16* __restrict__ dst,
                            int Nreal, int Kreal, int Npad, int Kpad, int zeroDiag) {
    int i = blockIdx.x * blockDim.x + threadIdx.x;
    int total = Npad * Kpad;
    if (i >= total) return;
    int n = i / Kpad, k = i % Kpad;
    float v = 0.f;
    if (n < Nreal && k < Kreal) {
        v = src[(size_t)n * Kreal + k];
        if (zeroDiag && n == k) v = 0.f;  // FeatureRegression off-diagonal mask
    }
    dst[i] = (_Float16)v;
}

// ---------------- WMMA fragment loaders ----------------
// 16-bit A/B fragment: halves[0..7] = K kb..kb+7, halves[8..15] = K kb+16..kb+23,
// kb = ks*32 + 8*(lane>>4); row (A: M, B: N) = lane&15.
static __device__ __forceinline__ v16h ldfrag(const _Float16* p) {
    v8h lo = *(const v8h*)p;
    v8h hi = *(const v8h*)(p + 16);
    return __builtin_shufflevector(lo, hi, 0,1,2,3,4,5,6,7,8,9,10,11,12,13,14,15);
}

static __device__ __forceinline__ float sigm(float x) { return 1.f / (1.f + expf(-x)); }

// out[16][ldo] (cols 0..ntiles*16-1) = A[16][K] @ W[Npad][K]^T (+ A2 @ W2^T) + b1 (+ b2)
static __device__ __forceinline__ void mm_tiles(
    const _Float16* __restrict__ A,  int lda,  int K,
    const _Float16* __restrict__ W,  int ldw,
    const _Float16* __restrict__ A2, int lda2, int Kb,
    const _Float16* __restrict__ W2, int ldw2,
    float* __restrict__ out, int ldo, int ntiles,
    const float* __restrict__ b1, const float* __restrict__ b2, int nreal)
{
    const int tid  = threadIdx.x;
    const int wave = tid >> 5;
    const int lane = tid & 31;
    const int l15  = lane & 15;
    const int rsel = lane >> 4;
    const int kb0  = rsel << 3;
    const int rbase = rsel << 3;  // D rows: M = v + 8*(lane>>4)

    for (int nt = wave; nt < ntiles; nt += NWAVES) {
        const int n = nt * 16 + l15;
        float b = 0.f;
        if (n < nreal) { b = b1[n]; if (b2) b += b2[n]; }
        v8f acc = { b, b, b, b, b, b, b, b };

        for (int ks = 0; ks < K / 32; ++ks) {
            const int kb = ks * 32 + kb0;
            v16h a = ldfrag(A + (size_t)l15 * lda + kb);
            v16h w = ldfrag(W + (size_t)n   * ldw + kb);
            acc = __builtin_amdgcn_wmma_f32_16x16x32_f16(
                false, a, false, w, (short)0, acc, false, false);
        }
        if (A2) {
            for (int ks = 0; ks < Kb / 32; ++ks) {
                const int kb = ks * 32 + kb0;
                v16h a = ldfrag(A2 + (size_t)l15 * lda2 + kb);
                v16h w = ldfrag(W2 + (size_t)n   * ldw2 + kb);
                acc = __builtin_amdgcn_wmma_f32_16x16x32_f16(
                    false, a, false, w, (short)0, acc, false, false);
            }
        }
        #pragma unroll
        for (int v = 0; v < 8; ++v)
            out[(size_t)(rbase + v) * ldo + nt * 16 + l15] = acc[v];
    }
}

// ---------------- main recurrent kernel ----------------
// One workgroup owns 16 batch rows for all 456 steps; state lives in LDS.
__global__ __launch_bounds__(NTHREADS) void rits_kernel(
    const float* __restrict__ values, const float* __restrict__ masks,
    const float* __restrict__ deltas,
    const float* __restrict__ Wdx, const float* __restrict__ bdx,
    const float* __restrict__ bdh, const float* __restrict__ bhr,
    const float* __restrict__ bfr, const float* __restrict__ bwc,
    const float* __restrict__ bih, const float* __restrict__ bhh,
    const float* __restrict__ bih2, const float* __restrict__ bhh2,
    const float* __restrict__ Wo,  const float* __restrict__ bo,
    const _Float16* __restrict__ Wdh16,  const _Float16* __restrict__ Whr16,
    const _Float16* __restrict__ Wfr16,  const _Float16* __restrict__ Wwc16,
    const _Float16* __restrict__ Wih16,  const _Float16* __restrict__ Whh16,
    const _Float16* __restrict__ Wih216, const _Float16* __restrict__ Whh216,
    float* __restrict__ part, float* __restrict__ predictions,
    float* __restrict__ imputations)
{
    // persistent per-row state (fp32)
    __shared__ __align__(16) float sh [MT * HID];
    __shared__ __align__(16) float sc [MT * HID];
    __shared__ __align__(16) float sh0[MT * HID];
    __shared__ __align__(16) float sc0[MT * HID];
    // this-step x / m
    __shared__ __align__(16) float sx [MT * FP];
    __shared__ __align__(16) float sm [MT * FP];
    // union region: {x_h | z_h | alpha | gamma_x} overlaid with LSTM gate buffer [16][704]
    __shared__ __align__(16) float sun[MT * G4];
    // f16 A-operand staging
    __shared__ __align__(16) _Float16 sA[MT * K2];   // concat operands (K<=352)
    __shared__ __align__(16) _Float16 sB[MT * KF];   // 192-wide operands (d, h, x_c, h0)
    __shared__ float sred[4];

    #define SXH(r,j) sun[0 * MT * FP + (r) * FP + (j)]
    #define SZH(r,j) sun[1 * MT * FP + (r) * FP + (j)]
    #define SAL(r,j) sun[2 * MT * FP + (r) * FP + (j)]
    #define SGX(r,j) sun[3 * MT * FP + (r) * FP + (j)]
    #define SG(r,j)  sun[(r) * G4 + (j)]

    const int tid = threadIdx.x;
    const int wg  = blockIdx.x;
    const int b0  = wg * MT;

    // init state
    for (int e = tid; e < MT * HID; e += NTHREADS) {
        sh[e] = 0.f; sc[e] = 0.f; sh0[e] = 0.f; sc0[e] = 0.f;
    }
    if (tid < 4) sred[tid] = 0.f;
    __syncthreads();

    for (int t = 0; t < SEQ; ++t) {
        // -- phase 0: load x, m, d; gamma_x; stage d as f16 A operand --
        for (int e = tid; e < MT * KF; e += NTHREADS) {
            int r = e / KF, j = e % KF;
            float dv = 0.f;
            if (j < FEAT) {
                size_t gi = ((size_t)(b0 + r) * SEQ + t) * FEAT + j;
                float xv = values[gi], mv = masks[gi];
                dv = deltas[gi];
                sx[r * FP + j] = xv;
                sm[r * FP + j] = mv;
                float wd = Wdx[(size_t)j * FEAT + j];          // diag-masked linear
                SGX(r, j) = expf(-fmaxf(dv * wd + bdx[j], 0.f));
            } else if (j < FP) {
                sx[r * FP + j] = 0.f; sm[r * FP + j] = 0.f; SGX(r, j) = 0.f;
            }
            sB[r * KF + j] = (_Float16)dv;
        }
        // prefetch next step's activation rows toward this WGP while the GEMMs run
        if (t + 1 < SEQ && tid < MT * 3) {
            int r = tid % MT, which = tid / MT;
            size_t gi = ((size_t)(b0 + r) * SEQ + (t + 1)) * FEAT;
            const float* p = (which == 0) ? values : (which == 1) ? masks : deltas;
            __builtin_prefetch(p + gi, 0, 1);
            __builtin_prefetch(p + gi + 128, 0, 1);
        }
        __syncthreads();

        // -- phase 1: gamma_h pre-activation = d @ Wdh^T + bdh -> SAL --
        mm_tiles(sB, KF, KF, Wdh16, KF, nullptr, 0, 0, nullptr, 0,
                 &SAL(0, 0), FP, FP / 16, bdh, nullptr, HID);
        __syncthreads();

        // -- phase 2: h *= exp(-relu(.)); stage h f16 --
        for (int e = tid; e < MT * KF; e += NTHREADS) {
            int r = e / KF, j = e % KF;
            _Float16 hv16 = (_Float16)0.f;
            if (j < HID) {
                float hv = sh[r * HID + j] * expf(-fmaxf(SAL(r, j), 0.f));
                sh[r * HID + j] = hv;
                hv16 = (_Float16)hv;
            }
            sB[r * KF + j] = hv16;
        }
        __syncthreads();

        // -- phase 3: x_h = h @ Whr^T + bhr -> SXH --
        mm_tiles(sB, KF, KH, Whr16, KF, nullptr, 0, 0, nullptr, 0,
                 &SXH(0, 0), FP, FP / 16, bhr, nullptr, FEAT);
        __syncthreads();

        // -- phase 4: loss1, msum, x_c (f16); build concat(gamma_x, m) in sA --
        {
            float l1 = 0.f, lm = 0.f;
            for (int e = tid; e < MT * KF; e += NTHREADS) {
                int r = e / KF, j = e % KF;
                _Float16 xc16 = (_Float16)0.f;
                if (j < FEAT) {
                    float m = sm[r * FP + j], x = sx[r * FP + j], xh = SXH(r, j);
                    float xc = m * x + (1.f - m) * xh;
                    xc16 = (_Float16)xc;
                    l1 += fabsf(x - xh) * m;
                    lm += m;
                }
                sB[r * KF + j] = xc16;
            }
            for (int e = tid; e < MT * K2; e += NTHREADS) {
                int r = e / K2, j = e % K2;
                _Float16 v = (_Float16)0.f;
                if (j < FEAT)          v = (_Float16)SGX(r, j);
                else if (j < 2 * FEAT) v = (_Float16)sm[r * FP + (j - FEAT)];
                sA[r * K2 + j] = v;
            }
            atomicAdd(&sred[0], l1);
            atomicAdd(&sred[3], lm);
        }
        __syncthreads();

        // -- phase 5: z_h = x_c @ Wfr_m^T + bfr -> SZH --
        mm_tiles(sB, KF, KF, Wfr16, KF, nullptr, 0, 0, nullptr, 0,
                 &SZH(0, 0), FP, FP / 16, bfr, nullptr, FEAT);
        // -- phase 6: alpha = [gamma_x, m] @ Wwc^T + bwc -> SAL --
        mm_tiles(sA, K2, K2, Wwc16, K2, nullptr, 0, 0, nullptr, 0,
                 &SAL(0, 0), FP, FP / 16, bwc, nullptr, FEAT);
        __syncthreads();

        // -- phase 7: c_h, c_c, loss2/3, imputation store; build concat(c_c, m) in sA; h0 f16 in sB --
        {
            float l2 = 0.f, l3 = 0.f;
            for (int e = tid; e < MT * K2; e += NTHREADS) {
                int r = e / K2, j = e % K2;
                _Float16 v = (_Float16)0.f;
                if (j < FEAT) {
                    float al = SAL(r, j), zh = SZH(r, j), xh = SXH(r, j);
                    float m = sm[r * FP + j], x = sx[r * FP + j];
                    float ch = al * zh + (1.f - al) * xh;
                    float cc = m * x + (1.f - m) * ch;
                    imputations[((size_t)(b0 + r) * SEQ + t) * FEAT + j] = cc;
                    v = (_Float16)cc;
                    l2 += fabsf(x - zh) * m;
                    l3 += fabsf(x - ch) * m;
                } else if (j < 2 * FEAT) {
                    v = (_Float16)sm[r * FP + (j - FEAT)];
                }
                sA[r * K2 + j] = v;
            }
            for (int e = tid; e < MT * KF; e += NTHREADS) {
                int r = e / KF, j = e % KF;
                sB[r * KF + j] = (j < HID) ? (_Float16)sh0[r * HID + j] : (_Float16)0.f;
            }
            atomicAdd(&sred[1], l2);
            atomicAdd(&sred[2], l3);
        }
        __syncthreads();
        if (tid < 4) {  // per-step partials for the cross-workgroup loss reduction
            part[((size_t)tid * SEQ + t) * NWG + wg] = sred[tid];
            sred[tid] = 0.f;
        }

        // -- phase 8: LSTM1 gates = [c_c,m] @ Wih^T + h0 @ Whh^T + bih + bhh -> SG --
        mm_tiles(sA, K2, K2, Wih16, K2, sB, KF, KH, Whh16, KF,
                 &SG(0, 0), G4, G4 / 16, bih, bhh, G4);
        __syncthreads();

        // -- phase 9: LSTM1 pointwise; stage h0 f16 (sB) and h f16 (sA, lda=K2) --
        for (int e = tid; e < MT * HID; e += NTHREADS) {
            int r = e / HID, j = e % HID;
            float gi = SG(r, j), gf = SG(r, j + HID), gg = SG(r, j + 2 * HID), go = SG(r, j + 3 * HID);
            float cn = sigm(gf) * sc0[r * HID + j] + sigm(gi) * tanhf(gg);
            float hn = sigm(go) * tanhf(cn);
            sc0[r * HID + j] = cn;
            sh0[r * HID + j] = hn;
            sB[r * KF + j] = (_Float16)hn;   // pad cols 176..191 still 0 from phase 7
        }
        for (int e = tid; e < MT * KF; e += NTHREADS) {
            int r = e / KF, j = e % KF;
            sA[r * K2 + j] = (j < HID) ? (_Float16)sh[r * HID + j] : (_Float16)0.f;
        }
        __syncthreads();

        // -- phase 10: LSTM2 gates = h0 @ Wih2^T + h @ Whh2^T + bih2 + bhh2 -> SG --
        mm_tiles(sB, KF, KH, Wih216, KF, sA, K2, KH, Whh216, KF,
                 &SG(0, 0), G4, G4 / 16, bih2, bhh2, G4);
        __syncthreads();

        // -- phase 11: LSTM2 pointwise: update h, c --
        for (int e = tid; e < MT * HID; e += NTHREADS) {
            int r = e / HID, j = e % HID;
            float gi = SG(r, j), gf = SG(r, j + HID), gg = SG(r, j + 2 * HID), go = SG(r, j + 3 * HID);
            float cn = sigm(gf) * sc[r * HID + j] + sigm(gi) * tanhf(gg);
            float hn = sigm(go) * tanhf(cn);
            sc[r * HID + j] = cn;
            sh[r * HID + j] = hn;
        }
        __syncthreads();
    }

    // predictions = sigmoid(h @ Wo^T + bo)
    if (tid < MT) {
        int row = tid;
        float acc = bo[0];
        for (int k = 0; k < HID; ++k) acc += sh[row * HID + k] * Wo[k];
        predictions[b0 + row] = sigm(acc);
    }
}

// ---------------- loss finalize: sum partials across workgroups per step ----------------
__global__ void finalize_loss(const float* __restrict__ part, float* __restrict__ out_loss) {
    __shared__ float red[256];
    float acc = 0.f;
    for (int t = threadIdx.x; t < SEQ; t += 256) {
        float a = 0.f, b = 0.f, c = 0.f, mm = 0.f;
        for (int w = 0; w < NWG; ++w) {
            a  += part[((size_t)0 * SEQ + t) * NWG + w];
            b  += part[((size_t)1 * SEQ + t) * NWG + w];
            c  += part[((size_t)2 * SEQ + t) * NWG + w];
            mm += part[((size_t)3 * SEQ + t) * NWG + w];
        }
        acc += (a + b + c) / (mm + 1e-5f);
    }
    red[threadIdx.x] = acc;
    __syncthreads();
    for (int s = 128; s > 0; s >>= 1) {
        if (threadIdx.x < s) red[threadIdx.x] += red[threadIdx.x + s];
        __syncthreads();
    }
    if (threadIdx.x == 0) out_loss[0] = red[0] / (float)SEQ;
}

// ---------------- host launch ----------------
extern "C" void kernel_launch(void* const* d_in, const int* in_sizes, int n_in,
                              void* d_out, int out_size, void* d_ws, size_t ws_size,
                              hipStream_t stream) {
    (void)in_sizes; (void)n_in; (void)out_size; (void)ws_size;
    const float* values = (const float*)d_in[0];
    const float* masks  = (const float*)d_in[1];
    const float* deltas = (const float*)d_in[2];
    // d_in[3]=labels, d_in[4]=is_train (unused by the returned outputs)
    const float* Wdh = (const float*)d_in[5];
    const float* bdh = (const float*)d_in[6];
    const float* Wdx = (const float*)d_in[7];
    const float* bdx = (const float*)d_in[8];
    const float* Whr = (const float*)d_in[9];
    const float* bhr = (const float*)d_in[10];
    const float* Wfr = (const float*)d_in[11];
    const float* bfr = (const float*)d_in[12];
    const float* Wwc = (const float*)d_in[13];
    const float* bwc = (const float*)d_in[14];
    const float* Wih = (const float*)d_in[15];
    const float* Whh = (const float*)d_in[16];
    const float* bih = (const float*)d_in[17];
    const float* bhh = (const float*)d_in[18];
    const float* Wih2 = (const float*)d_in[19];
    const float* Whh2 = (const float*)d_in[20];
    const float* bih2 = (const float*)d_in[21];
    const float* bhh2 = (const float*)d_in[22];
    const float* Wo  = (const float*)d_in[23];
    const float* bo  = (const float*)d_in[24];

    char* ws = (char*)d_ws;
    _Float16* Wdh16   = (_Float16*)(ws + OFF_WDH);
    _Float16* Whr16   = (_Float16*)(ws + OFF_WHR);
    _Float16* Wfr16   = (_Float16*)(ws + OFF_WFR);
    _Float16* Wwc16   = (_Float16*)(ws + OFF_WWC);
    _Float16* Wih16   = (_Float16*)(ws + OFF_WIH);
    _Float16* Whh16   = (_Float16*)(ws + OFF_WHH);
    _Float16* Wih216  = (_Float16*)(ws + OFF_WIH2);
    _Float16* Whh216  = (_Float16*)(ws + OFF_WHH2);
    float*    part    = (float*)(ws + OFF_PART);

    auto cvt = [&](const float* src, _Float16* dst, int Nr, int Kr, int Np, int Kp, int zd) {
        int total = Np * Kp;
        cvt_pad_f16<<<(total + 255) / 256, 256, 0, stream>>>(src, dst, Nr, Kr, Np, Kp, zd);
    };
    cvt(Wdh,  Wdh16,  HID,  FEAT,     FP, KF, 0);
    cvt(Whr,  Whr16,  FEAT, HID,      FP, KF, 0);
    cvt(Wfr,  Wfr16,  FEAT, FEAT,     FP, KF, 1);   // off-diagonal mask
    cvt(Wwc,  Wwc16,  FEAT, 2 * FEAT, FP, K2, 0);
    cvt(Wih,  Wih16,  G4,   2 * FEAT, G4, K2, 0);
    cvt(Whh,  Whh16,  G4,   HID,      G4, KF, 0);
    cvt(Wih2, Wih216, G4,   HID,      G4, KF, 0);
    cvt(Whh2, Whh216, G4,   HID,      G4, KF, 0);

    float* out = (float*)d_out;
    rits_kernel<<<NWG, NTHREADS, 0, stream>>>(
        values, masks, deltas,
        Wdx, bdx, bdh, bhr, bfr, bwc, bih, bhh, bih2, bhh2, Wo, bo,
        Wdh16, Whr16, Wfr16, Wwc16, Wih16, Whh16, Wih216, Whh216,
        part, out + 1, out + 1 + BATCH);

    finalize_loss<<<1, 256, 0, stream>>>(part, out);
}
